// LIFLayer_81226421502660
// MI455X (gfx1250) — matrix operand order repaired and checked by reading, not compiled, not run on hardware
//
#include <hip/hip_runtime.h>

typedef __attribute__((ext_vector_type(16))) __bf16 v16bf;
typedef __attribute__((ext_vector_type(8)))  __bf16 v8bf;
typedef __attribute__((ext_vector_type(4)))  __bf16 v4bf;
typedef __attribute__((ext_vector_type(8)))  float  v8f;

#define T_STEPS 200
#define BATCH   256
#define IN_F    512
#define OUT_F   512
#define BT      16   // batch rows per workgroup
#define OT      32   // out cols per workgroup (2 waves x 16)

// 2 waves / 64 threads per block, 256 persistent workgroups.
// Each wave computes one 16x16 WMMA C tile; membrane potential v lives in
// the accumulator-register layout across the whole T loop (full fusion: the
// 105MB 'cur' intermediate never touches HBM). sX is double-buffered so the
// staging of x[t+1] (global load + fp32->bf16 cvt + ds_store) overlaps the
// 16 WMMAs consuming x[t]; one barrier per timestep.
__device__ __forceinline__ void stage_x(const float* __restrict__ xt,
                                        __bf16* __restrict__ dst,
                                        int tid, bool pf_next) {
  for (int i = tid; i < (BT * IN_F) / 4; i += 64) {
    const int e = i * 4;
    const int r = e >> 9;            // /512
    const int c = e & (IN_F - 1);
    float4 f = *(const float4*)(xt + (size_t)r * IN_F + c);
    v4bf h;
    h[0] = (__bf16)f.x; h[1] = (__bf16)f.y;
    h[2] = (__bf16)f.z; h[3] = (__bf16)f.w;
    *(v4bf*)(&dst[e]) = h;
    if (pf_next && ((c & 31) == 0))  // one global_prefetch_b8 per 128B, t+2
      __builtin_prefetch(xt + (size_t)BATCH * IN_F + (size_t)r * IN_F + c, 0, 1);
  }
}

__launch_bounds__(64, 1)
__global__ void lif_fused_wmma(const float* __restrict__ x,
                               const float* __restrict__ W,
                               const float* __restrict__ bias,
                               float* __restrict__ out) {
  __shared__ __bf16 sW[OT * IN_F];       // 32 KB: W rows o0..o0+31, bf16
  __shared__ __bf16 sX[2][BT * IN_F];    // 2 x 16 KB: double-buffered x[t]

  const int tid   = threadIdx.x;   // 0..63
  const int wave  = tid >> 5;      // 0..1
  const int lane  = tid & 31;
  const int half  = lane >> 4;     // 0/1 lane group
  const int row   = lane & 15;
  const int nBase = wave * 16;     // waves split the OUT tile (x reused 2x)
  const int b0 = blockIdx.y * BT;
  const int o0 = blockIdx.x * OT;

  // ---- Stage W tile once: fp32 -> bf16 into LDS ----
  for (int i = tid; i < (OT * IN_F) / 4; i += 64) {
    const int e = i * 4;
    const int r = e >> 9;
    const int c = e & (IN_F - 1);
    float4 f = *(const float4*)(W + (size_t)(o0 + r) * IN_F + c);
    v4bf h;
    h[0] = (__bf16)f.x; h[1] = (__bf16)f.y;
    h[2] = (__bf16)f.z; h[3] = (__bf16)f.w;
    *(v4bf*)(&sW[e]) = h;
  }

  // ---- Prologue: stage x[0] into buffer 0 ----
  stage_x(x + (size_t)b0 * IN_F, sX[0], tid, true);

  const float cb = bias[o0 + nBase + row];              // bias for lane's N col
  v8f vmem = {0.f, 0.f, 0.f, 0.f, 0.f, 0.f, 0.f, 0.f}; // LIF membrane state
  __syncthreads();

  for (int t = 0; t < T_STEPS; ++t) {
    const int cur = t & 1;

    // ---- Stage x[t+1] into the other buffer (overlaps with WMMAs below) ----
    if (t + 1 < T_STEPS)
      stage_x(x + ((size_t)(t + 1) * BATCH + b0) * IN_F, sX[cur ^ 1], tid,
              (t + 2) < T_STEPS);

    // ---- 16x16 tile GEMM: C = x_tile @ W_tileT + b, K = 512 ----
    v8f acc = {cb, cb, cb, cb, cb, cb, cb, cb};
    const __bf16* aRow = &sX[cur][row * IN_F];         // A: row M = row
    const __bf16* bRow = &sW[(nBase + row) * IN_F];    // B: col N = row
#pragma unroll
    for (int kc = 0; kc < IN_F; kc += 32) {
      // A 16x32 bf16 layout: VGPR0-3 K = half*8+0..7, VGPR4-7 K = 16+half*8+0..7
      union { v16bf v; v8bf h[2]; } a;
      a.h[0] = *(const v8bf*)(aRow + kc + half * 8);
      a.h[1] = *(const v8bf*)(aRow + kc + 16 + half * 8);
      // B 32x16 bf16 layout: same K order as A (half*8+0..7, 16+half*8+0..7)
      union { v16bf v; v8bf h[2]; } b;
      b.h[0] = *(const v8bf*)(bRow + kc + half * 8);
      b.h[1] = *(const v8bf*)(bRow + kc + 16 + half * 8);
      v16bf bf = b.v;
      acc = __builtin_amdgcn_wmma_f32_16x16x32_bf16(
          /*neg_a=*/false, a.v, /*neg_b=*/false, bf,
          /*c_mod=*/(short)0, acc, /*reuse_a=*/false, /*reuse_b=*/false);
    }

    // ---- LIF update on accumulator (C/D layout: M = r8 + half*8, N = row) ----
    float* op = out + (size_t)t * BATCH * OUT_F
                    + (size_t)(b0 + half * 8) * OUT_F
                    + (o0 + nBase + row);
#pragma unroll
    for (int r8 = 0; r8 < 8; ++r8) {
      float v = vmem[r8];
      v += (acc[r8] - v) * 0.1f;            // v += (cur - v)/TAU, TAU=10
      float s = (v > 1.0f) ? 1.0f : 0.0f;   // spike(v - V_TH), V_TH=1
      v *= (1.0f - s);                      // reset (detached)
      vmem[r8] = v;
      op[(size_t)r8 * OUT_F] = s;
    }
    __syncthreads();  // staging(t+1) done before it is consumed; compute(t)
                      // done before buffer 'cur' is overwritten at t+1
  }
}

extern "C" void kernel_launch(void* const* d_in, const int* in_sizes, int n_in,
                              void* d_out, int out_size, void* d_ws, size_t ws_size,
                              hipStream_t stream) {
  const float* x = (const float*)d_in[0];   // [T, B, IN] fp32
  const float* W = (const float*)d_in[1];   // [OUT, IN] fp32
  const float* b = (const float*)d_in[2];   // [OUT] fp32
  float* out = (float*)d_out;               // [T, B, OUT] fp32 spikes
  (void)in_sizes; (void)n_in; (void)out_size; (void)d_ws; (void)ws_size;

  dim3 grid(OUT_F / OT, BATCH / BT);        // (16, 16) = 256 workgroups
  dim3 block(64);                           // 2 wave32s
  lif_fused_wmma<<<grid, block, 0, stream>>>(x, W, b, out);
}
